// MnistNet_47253230190811
// MI455X (gfx1250) — matrix operand order repaired and checked by reading, not compiled, not run on hardware
//
#include <hip/hip_runtime.h>

typedef __attribute__((ext_vector_type(16))) _Float16 v16h;
typedef __attribute__((ext_vector_type(8)))  _Float16 v8h;
typedef __attribute__((ext_vector_type(4)))  _Float16 v4h;
typedef __attribute__((ext_vector_type(8)))  float    v8f;

// exact pointee type of the async-to-LDS builtin's pointer params
typedef int vi4 __attribute__((vector_size(4 * sizeof(int))));
typedef __attribute__((address_space(1))) vi4* gvi4p;   // global
typedef __attribute__((address_space(3))) vi4* lvi4p;   // LDS

#define BM 128
#define BN 64
#define BK 32
#define NT 256    // 8 waves
#define LDST 40   // padded LDS row stride in halves (80B) -> conflict-free 16B reads

#if __has_builtin(__builtin_amdgcn_global_load_async_to_lds_b128)
#define HAVE_ASYNC_LDS 1
#else
#define HAVE_ASYNC_LDS 0
#endif

__device__ __forceinline__ void wait_asynccnt0() {
#if __has_builtin(__builtin_amdgcn_s_wait_asynccnt)
  __builtin_amdgcn_s_wait_asynccnt(0);
#else
  asm volatile("s_wait_asynccnt 0x0" ::: "memory");
#endif
}

// C[M, ldc] = A[M,K](f32) @ Bw[Npad,Kpad](f16 row-major = W, i.e. B^T), f32 accumulate.
// A converted f32->f16 while staging to LDS (VGPR path); B tile copied with async
// global->LDS (ASYNCcnt) when available. Double-buffered LDS, branchless guards.
__global__ __launch_bounds__(NT) void snn_gemm_wmma(
    const float* __restrict__ A,
    const _Float16* __restrict__ Bw,
    float* __restrict__ C,
    int M, int K, int Kpad, int Npad, int Nout, int ldc)
{
  __shared__ _Float16 As[2][BM * LDST];
  __shared__ _Float16 Bs[2][BN * LDST];

  const int tid  = threadIdx.x;
  const int lane = tid & 31;
  const int wave = tid >> 5;    // 0..7
  const int wm   = wave >> 1;   // 0..3  -> 32-row strip
  const int wn   = wave & 1;    // 0..1  -> 32-col strip
  const int lr   = lane & 15;
  const int hi   = lane >> 4;

  const int mBase = blockIdx.y * BM;
  const int nBase = blockIdx.x * BN;

  v8f acc[2][2];
  {
    v8f z = {};
    acc[0][0] = z; acc[0][1] = z; acc[1][0] = z; acc[1][1] = z;
  }

  const int numK = Kpad / BK;

  float4 aReg[4];   // 128x32 f32 tile, 4 float4 per thread
#if !HAVE_ASYNC_LDS
  uint4  bReg;      // 64x32 f16 tile, 8 halves per thread
#endif

  // ---- stage tile kt: A -> registers (cvt later), B -> LDS[dstBuf] ----
  auto loadTile = [&](int kt, int dstBuf) {
    const int k0 = kt * BK;
    #pragma unroll
    for (int j = 0; j < 4; ++j) {
      const int idx4 = tid + NT * j;          // 0..1023 (8 float4 per 32-float row)
      const int row  = idx4 >> 3;
      const int c4   = idx4 & 7;
      const int kg   = k0 + c4 * 4;
      const int kc   = (kg < K) ? kg : 0;     // clamp: always a valid address
      float4 v = *(const float4*)(A + (size_t)(mBase + row) * K + kc);
      const bool ok = (kg < K);
      v.x = ok ? v.x : 0.0f;
      v.y = ok ? v.y : 0.0f;
      v.z = ok ? v.z : 0.0f;
      v.w = ok ? v.w : 0.0f;
      aReg[j] = v;
    }
    {
      const int row = tid >> 2;               // 4 x 8-half chunks per 32-half row
      const int c8  = tid & 3;
      const int n   = nBase + row;
      // Clamp to last (zero-padded) weight row: columns >= Npad are never stored
      // (Nout <= Npad), and padded rows are zeros, so no zero-fill is needed.
      const int nc  = (n < Npad) ? n : (Npad - 1);
      const _Float16* src = Bw + (size_t)nc * Kpad + k0 + c8 * 8;
      _Float16* dst = &Bs[dstBuf][row * LDST + c8 * 8];
#if HAVE_ASYNC_LDS
      __builtin_amdgcn_global_load_async_to_lds_b128(
          (gvi4p)(vi4*)src, (lvi4p)(vi4*)dst, 0, 0);
#else
      (void)dst;
      uint4 b = *(const uint4*)src;
      const bool ok = (n < Npad);
      b.x = ok ? b.x : 0u;
      b.y = ok ? b.y : 0u;
      b.z = ok ? b.z : 0u;
      b.w = ok ? b.w : 0u;
      bReg = b;
#endif
    }
  };

  // ---- register -> LDS staging (A converted to f16; B only on fallback path) ----
  auto storeTile = [&](int buf) {
    #pragma unroll
    for (int j = 0; j < 4; ++j) {
      const int idx4 = tid + NT * j;
      const int row  = idx4 >> 3;
      const int c4   = idx4 & 7;
      v4h h;
      h[0] = (_Float16)aReg[j].x; h[1] = (_Float16)aReg[j].y;
      h[2] = (_Float16)aReg[j].z; h[3] = (_Float16)aReg[j].w;
      *(v4h*)(&As[buf][row * LDST + c4 * 4]) = h;
    }
#if !HAVE_ASYNC_LDS
    {
      const int row = tid >> 2;
      const int c8  = tid & 3;
      *(uint4*)(&Bs[buf][row * LDST + c8 * 8]) = bReg;
    }
#endif
  };

  // prologue: stage tile 0
  loadTile(0, 0);
  storeTile(0);
#if HAVE_ASYNC_LDS
  wait_asynccnt0();
#endif
  __syncthreads();

  for (int kt = 0; kt < numK; ++kt) {
    const int  cur  = kt & 1;
    const bool more = (kt + 1) < numK;

    if (more) loadTile(kt + 1, cur ^ 1);  // issue next tile's loads early

    // ---- fragments per ISA layouts ----
    // A 16x32: lanes 0-15 (row=lr): K {0..7,16..23}; lanes 16-31: K {8..15,24..31}
    v16h aF[2];
    #pragma unroll
    for (int t = 0; t < 2; ++t) {
      const int rowA = wm * 32 + t * 16 + lr;
      v8h lo = *(const v8h*)(&As[cur][rowA * LDST + hi * 8]);
      v8h hh = *(const v8h*)(&As[cur][rowA * LDST + hi * 8 + 16]);
      aF[t] = __builtin_shufflevector(lo, hh, 0,1,2,3,4,5,6,7,8,9,10,11,12,13,14,15);
    }
    // B 32x16: lane = column; lanes 0-15 hold K {0..7,16..23}, lanes 16-31 hold K {8..15,24..31}
    v16h bF[2];
    #pragma unroll
    for (int t = 0; t < 2; ++t) {
      const int colB = wn * 32 + t * 16 + lr;
      v8h lo = *(const v8h*)(&Bs[cur][colB * LDST + hi * 8]);
      v8h hh = *(const v8h*)(&Bs[cur][colB * LDST + hi * 8 + 16]);
      bF[t] = __builtin_shufflevector(lo, hh, 0,1,2,3,4,5,6,7,8,9,10,11,12,13,14,15);
    }

    #pragma unroll
    for (int i = 0; i < 2; ++i)
      #pragma unroll
      for (int j = 0; j < 2; ++j)
        acc[i][j] = __builtin_amdgcn_wmma_f32_16x16x32_f16(
            false, aF[i], false, bF[j], (short)0, acc[i][j], false, false);

    if (more) storeTile(cur ^ 1); // fill the other buffer (A path)
#if HAVE_ASYNC_LDS
    wait_asynccnt0();             // own async B copies done; barrier makes all visible
#endif
    __syncthreads();              // single barrier per K-iteration
  }

  // ---- epilogue: C/D layout -> lane lr = column, VGPR r -> row (r + 8*hi) ----
  #pragma unroll
  for (int ti = 0; ti < 2; ++ti)
    #pragma unroll
    for (int tj = 0; tj < 2; ++tj) {
      const int col = nBase + wn * 32 + tj * 16 + lr;
      if (col < Nout) {
        const int rbase = mBase + wm * 32 + ti * 16 + hi * 8;
        #pragma unroll
        for (int r = 0; r < 8; ++r)
          C[(size_t)(rbase + r) * ldc + col] = acc[ti][tj][r];
      }
    }
}

// f32 weights [Nsrc,K] -> zero-padded f16 [Npad,Kpad]
__global__ void convert_weights(const float* __restrict__ W, _Float16* __restrict__ Wh,
                                int Nsrc, int K, int Npad, int Kpad)
{
  const int idx = blockIdx.x * blockDim.x + threadIdx.x;
  if (idx >= Npad * Kpad) return;
  const int n = idx / Kpad;
  const int k = idx - n * Kpad;
  const float v = (n < Nsrc && k < K) ? W[(size_t)n * K + k] : 0.0f;
  Wh[idx] = (_Float16)v;
}

// LIF scan: mem region currently holds cur[t]; overwrite in place with mem[t],
// write spk[t] to spk region. One thread per (batch, neuron).
__global__ void lif_scan(float* __restrict__ out, size_t spkBase, size_t memBase, int S)
{
  const int i = blockIdx.x * blockDim.x + threadIdx.x;
  if (i >= S) return;
  float mem = 0.0f;
  for (int t = 0; t < 50; ++t) {
    const size_t off = (size_t)t * (size_t)S + (size_t)i;
    const float cur   = out[memBase + off];
    const float reset = (mem > 1.0f) ? 1.0f : 0.0f;
    mem = 0.9f * mem + cur - reset;           // beta*mem + cur - reset*thresh
    out[memBase + off] = mem;
    out[spkBase + off] = (mem > 1.0f) ? 1.0f : 0.0f;
  }
}

extern "C" void kernel_launch(void* const* d_in, const int* in_sizes, int n_in,
                              void* d_out, int out_size, void* d_ws, size_t ws_size,
                              hipStream_t stream)
{
  (void)in_sizes; (void)n_in; (void)out_size; (void)ws_size;

  const float* x  = (const float*)d_in[0];   // [50*1024, 784]
  const float* W0 = (const float*)d_in[1];   // [1024, 784]
  const float* W1 = (const float*)d_in[2];   // [1024, 1024]
  const float* W2 = (const float*)d_in[3];   // [10, 1024]
  float* out = (float*)d_out;

  _Float16* W0h = (_Float16*)d_ws;           // [1024, 800]
  _Float16* W1h = W0h + 1024 * 800;          // [1024, 1024]
  _Float16* W2h = W1h + 1024 * 1024;         // [16, 1024]

  // d_out layout (floats): spk0, spk1, spk2, mem0, mem1, mem2
  const size_t spk0 = 0;
  const size_t spk1 = 52428800;
  const size_t spk2 = 104857600;
  const size_t mem0 = 105369600;
  const size_t mem1 = 157798400;
  const size_t mem2 = 210227200;

  convert_weights<<<(1024*800  + 255)/256, 256, 0, stream>>>(W0, W0h, 1024,  784, 1024,  800);
  convert_weights<<<(1024*1024 + 255)/256, 256, 0, stream>>>(W1, W1h, 1024, 1024, 1024, 1024);
  convert_weights<<<(16*1024   + 255)/256, 256, 0, stream>>>(W2, W2h,   10, 1024,   16, 1024);

  const int M = 51200;                       // 50 steps x 1024 batch
  dim3 blk(NT);

  // cur0 = x @ W0^T  -> mem0 region (f32)
  dim3 g1(1024 / BN, M / BM);
  snn_gemm_wmma<<<g1, blk, 0, stream>>>(x, W0h, out + mem0, M, 784, 800, 1024, 1024, 1024);
  // cur1 = cur0 @ W1^T -> mem1 region
  snn_gemm_wmma<<<g1, blk, 0, stream>>>(out + mem0, W1h, out + mem1, M, 1024, 1024, 1024, 1024, 1024);
  // cur2 = cur1 @ W2^T -> mem2 region (ldc=10, only cols 0..9 stored)
  dim3 g3(1, M / BM);
  snn_gemm_wmma<<<g3, blk, 0, stream>>>(out + mem1, W2h, out + mem2, M, 1024, 1024, 16, 10, 10);

  // LIF scans (after ALL gemms: gemm3 reads mem1 region before scan overwrites it)
  lif_scan<<<(1024*1024 + 255)/256, 256, 0, stream>>>(out, spk0, mem0, 1024 * 1024);
  lif_scan<<<(1024*1024 + 255)/256, 256, 0, stream>>>(out, spk1, mem1, 1024 * 1024);
  lif_scan<<<(10240     + 255)/256, 256, 0, stream>>>(out, spk2, mem2, 10240);
}
